// RelativeAttention_60198261621496
// MI455X (gfx1250) — matrix operand order repaired and checked by reading, not compiled, not run on hardware
//
#include <hip/hip_runtime.h>
#include <hip/hip_bf16.h>

#define BDIM   2
#define NSEQ   2048
#define DMODEL 512
#define NHEADS 8
#define DHEAD  64
#define MTOK   (BDIM * NSEQ)   // 4096 tokens

typedef __attribute__((ext_vector_type(16))) __bf16 bf16x16;
typedef __attribute__((ext_vector_type(8)))  __bf16 bf16x8;
typedef __attribute__((ext_vector_type(8)))  float  f32x8;

__device__ __forceinline__ __bf16 f2bf(float f) {
    unsigned u; __builtin_memcpy(&u, &f, 4);
    unsigned r = u + 0x7FFFu + ((u >> 16) & 1u);   // round-to-nearest-even
    unsigned short h = (unsigned short)(r >> 16);
    __bf16 b; __builtin_memcpy(&b, &h, 2);
    return b;
}

__device__ __forceinline__ f32x8 zero8() {
    f32x8 z;
#pragma unroll
    for (int i = 0; i < 8; ++i) z[i] = 0.f;
    return z;
}

// Build a 16-element bf16 fragment from two contiguous 16-byte chunks.
__device__ __forceinline__ bf16x16 load_frag(const __bf16* p0, const __bf16* p1) {
    bf16x8 a = *(const bf16x8*)p0;
    bf16x8 b = *(const bf16x8*)p1;
    bf16x16 r;
#pragma unroll
    for (int i = 0; i < 8; ++i) { r[i] = a[i]; r[i + 8] = b[i]; }
    return r;
}

__device__ __forceinline__ f32x8 wmma_bf16(bf16x16 a, bf16x16 b, f32x8 c) {
    // 8 args: (neg_a, A, neg_b, B, c_mod, C, reuse_a, reuse_b)
    return __builtin_amdgcn_wmma_f32_16x16x32_bf16(false, a, false, b, (short)0, c,
                                                   false, false);
}

// Row reductions across a 16-lane half of the wave32 (C-matrix rows live in
// VGPRs, columns across 16 lanes; lane halves hold rows 0-7 / 8-15).
__device__ __forceinline__ float rowmax16(float v) {
#pragma unroll
    for (int off = 8; off >= 1; off >>= 1) v = fmaxf(v, __shfl_xor(v, off, 32));
    return v;
}
__device__ __forceinline__ float rowsum16(float v) {
#pragma unroll
    for (int off = 8; off >= 1; off >>= 1) v += __shfl_xor(v, off, 32);
    return v;
}

// ---------------------------------------------------------------- converts
__global__ __launch_bounds__(256) void conv_kernel(const float* __restrict__ src,
                                                   __bf16* __restrict__ dst, int n) {
    for (int i = blockIdx.x * blockDim.x + threadIdx.x; i < n;
         i += gridDim.x * blockDim.x)
        dst[i] = f2bf(src[i]);
}

// Transpose-convert a 512x512 weight: Wt[n][k] = W[k][n]  (blockIdx.z picks one)
__global__ __launch_bounds__(256) void convT_kernel(
    const float* __restrict__ s0, const float* __restrict__ s1,
    const float* __restrict__ s2, const float* __restrict__ s3,
    __bf16* __restrict__ d0, __bf16* __restrict__ d1,
    __bf16* __restrict__ d2, __bf16* __restrict__ d3) {
    const float* s = (blockIdx.z == 0) ? s0 : (blockIdx.z == 1) ? s1
                   : (blockIdx.z == 2) ? s2 : s3;
    __bf16* d = (blockIdx.z == 0) ? d0 : (blockIdx.z == 1) ? d1
              : (blockIdx.z == 2) ? d2 : d3;
    int kx = blockIdx.x * 16 + (threadIdx.x & 15);
    int ny = blockIdx.y * 16 + (threadIdx.x >> 4);
    d[(size_t)ny * DMODEL + kx] = f2bf(s[(size_t)kx * DMODEL + ny]);
}

// ---------------------------------------------------------------- QKV GEMM
// out = x[4096,512] @ W[512,512]; W pre-transposed so each lane's B-fragment
// is a contiguous 32B run.  32x32 tile per wave: 4 WMMAs per k-step from
// 8 b128 loads (2 loads per WMMA vs 4 for a 16x16 tile).
// blockIdx.z: 0=Q, 1=K, 2=V(transposed per-head layout).
__global__ __launch_bounds__(256) void qkv_gemm_kernel(
    const __bf16* __restrict__ xb, const __bf16* __restrict__ wqT,
    const __bf16* __restrict__ wkT, const __bf16* __restrict__ wvT,
    __bf16* __restrict__ qg, __bf16* __restrict__ kg, __bf16* __restrict__ vtg) {
    const int lane = threadIdx.x & 31, half = lane >> 4, lr = lane & 15;
    const int wave = threadIdx.x >> 5;
    const int mt = blockIdx.x;                 // 0..127 (32-row tiles)
    const int nt = blockIdx.y * 8 + wave;      // 0..15  (32-col tiles)
    const int mode = blockIdx.z;
    const __bf16* wT = (mode == 0) ? wqT : (mode == 1) ? wkT : wvT;

    const __bf16* ar0 = xb + (size_t)(mt * 32 + lr) * DMODEL;
    const __bf16* ar1 = ar0 + (size_t)16 * DMODEL;
    const __bf16* br0 = wT + (size_t)(nt * 32 + lr) * DMODEL;
    const __bf16* br1 = br0 + (size_t)16 * DMODEL;

    f32x8 c00 = zero8(), c01 = zero8(), c10 = zero8(), c11 = zero8();
#pragma unroll
    for (int ks = 0; ks < DMODEL / 32; ++ks) {
        const int dd = ks * 32;
        bf16x16 a0 = load_frag(ar0 + dd + half * 16, ar0 + dd + half * 16 + 8);
        bf16x16 a1 = load_frag(ar1 + dd + half * 16, ar1 + dd + half * 16 + 8);
        bf16x16 b0 = load_frag(br0 + dd + half * 16, br0 + dd + half * 16 + 8);
        bf16x16 b1 = load_frag(br1 + dd + half * 16, br1 + dd + half * 16 + 8);
        c00 = wmma_bf16(a0, b0, c00);
        c01 = wmma_bf16(a0, b1, c01);
        c10 = wmma_bf16(a1, b0, c10);
        c11 = wmma_bf16(a1, b1, c11);
    }
    // C layout per tile: col = lr, rows = half*8 + r
    f32x8 cs[2][2] = {{c00, c01}, {c10, c11}};
#pragma unroll
    for (int mi = 0; mi < 2; ++mi) {
#pragma unroll
        for (int ni = 0; ni < 2; ++ni) {
            const int f = nt * 32 + ni * 16 + lr;      // output feature
            const int hh = f >> 6, dd = f & 63;
#pragma unroll
            for (int r = 0; r < 8; ++r) {
                const int m = mt * 32 + mi * 16 + half * 8 + r;   // token id
                const int b = m >> 11, nrow = m & (NSEQ - 1);
                const int bh = b * NHEADS + hh;
                const __bf16 val = f2bf(cs[mi][ni][r]);
                if (mode == 2)      vtg[((size_t)bh * DHEAD + dd) * NSEQ + nrow] = val;
                else if (mode == 1) kg[((size_t)bh * NSEQ + nrow) * DHEAD + dd] = val;
                else                qg[((size_t)bh * NSEQ + nrow) * DHEAD + dd] = val;
            }
        }
    }
}

// ---------------------------------------------------------------- attention
// One wave per (b,h, 16 query rows).  Flash softmax over causal keys.
// Srel[i,j] = q_i . Er[N-1-(i-j)]  (exact skew identity, j<=i).
__global__ __launch_bounds__(32) void attn_kernel(
    const __bf16* __restrict__ qg, const __bf16* __restrict__ kg,
    const __bf16* __restrict__ vtg, const __bf16* __restrict__ erb,
    __bf16* __restrict__ ao) {
    __shared__ float  Gl[16][32];
    __shared__ __bf16 Pl[16][32];

    const int lane = threadIdx.x & 31, half = lane >> 4, lr = lane & 15;
    const int i0 = blockIdx.x * 16;
    const int bh = blockIdx.y;
    const __bf16* qb = qg + (size_t)bh * NSEQ * DHEAD;
    const __bf16* kb = kg + (size_t)bh * NSEQ * DHEAD;
    const __bf16* vb = vtg + (size_t)bh * DHEAD * NSEQ;

    // Q block as two A-fragments (dh=64 -> K steps of 32), resident all loop.
    bf16x16 qa[2];
    {
        const __bf16* qr = qb + (size_t)(i0 + lr) * DHEAD;
#pragma unroll
        for (int s = 0; s < 2; ++s) {
            const int d0 = s * 32 + half * 16;
            qa[s] = load_frag(qr + d0, qr + d0 + 8);
        }
    }

    f32x8 o[4];
#pragma unroll
    for (int t = 0; t < 4; ++t) o[t] = zero8();
    float mrow[8], lrow[8];
#pragma unroll
    for (int r = 0; r < 8; ++r) { mrow[r] = -3.0e38f; lrow[r] = 0.f; }

    const int njt = i0 / 16 + 1;                 // causal 16-col tiles
    for (int jt = 0; jt < njt; jt += 2) {
        f32x8 sacc[2];
        float tmax[8];
#pragma unroll
        for (int r = 0; r < 8; ++r) tmax[r] = -3.0e38f;

        for (int t = 0; t < 2; ++t) {
            const int jts = jt + t;
            f32x8 c = zero8();
            if (jts < njt) {
                const int j0 = jts * 16;
                // ---- QK^T tile (2 WMMA)
                const __bf16* kr = kb + (size_t)(j0 + lr) * DHEAD;
                {   // prefetch next tile's K row
                    int pj = j0 + 32 + lr; if (pj > NSEQ - 1) pj = NSEQ - 1;
                    __builtin_prefetch((const void*)(kb + (size_t)pj * DHEAD), 0, 1);
                }
#pragma unroll
                for (int s = 0; s < 2; ++s) {
                    bf16x16 b = load_frag(kr + s * 32 + half * 16,
                                          kr + s * 32 + half * 16 + 8);
                    c = wmma_bf16(qa[s], b, c);
                }
                // ---- G[ii,cc] = Q . Er[base0+cc], cc in [0,32)  (4 WMMA)
                const int base0 = NSEQ - 16 - i0 + j0;   // >= 0 always
                int r0 = base0 + lr;       if (r0 > NSEQ - 1) r0 = NSEQ - 1;
                int r1 = base0 + 16 + lr;  if (r1 > NSEQ - 1) r1 = NSEQ - 1;
                const __bf16* e0 = erb + (size_t)r0 * DHEAD;
                const __bf16* e1 = erb + (size_t)r1 * DHEAD;
                f32x8 g0 = zero8(), g1 = zero8();
#pragma unroll
                for (int s = 0; s < 2; ++s) {
                    g0 = wmma_bf16(qa[s], load_frag(e0 + s * 32 + half * 16,
                                                    e0 + s * 32 + half * 16 + 8), g0);
                    g1 = wmma_bf16(qa[s], load_frag(e1 + s * 32 + half * 16,
                                                    e1 + s * 32 + half * 16 + 8), g1);
                }
#pragma unroll
                for (int r = 0; r < 8; ++r) {
                    Gl[half * 8 + r][lr] = g0[r];
                    Gl[half * 8 + r][16 + lr] = g1[r];
                }
                __syncthreads();   // ds store -> ds load within wave
                // ---- scores: (QK + Srel) * 1/sqrt(dh), causal mask
#pragma unroll
                for (int r = 0; r < 8; ++r) {
                    const int ii = half * 8 + r;
                    const float srel = Gl[ii][15 - ii + lr];
                    float val = (c[r] + srel) * 0.125f;
                    if (j0 + lr > i0 + ii) val = -3.0e38f;
                    c[r] = val;
                    tmax[r] = fmaxf(tmax[r], val);
                }
                __syncthreads();
            } else {
#pragma unroll
                for (int r = 0; r < 8; ++r) c[r] = -3.0e38f;
            }
            sacc[t] = c;
        }

        // ---- online softmax update over the 32-key step
        float mnew[8], scl[8];
#pragma unroll
        for (int r = 0; r < 8; ++r) {
            const float tm = rowmax16(tmax[r]);
            mnew[r] = fmaxf(mrow[r], tm);
            scl[r] = __expf(mrow[r] - mnew[r]);
            mrow[r] = mnew[r];
        }
        float pvv[2][8];
#pragma unroll
        for (int t = 0; t < 2; ++t)
#pragma unroll
            for (int r = 0; r < 8; ++r) {
                const float pv = __expf(sacc[t][r] - mnew[r]);
                pvv[t][r] = pv;
                Pl[half * 8 + r][t * 16 + lr] = f2bf(pv);
            }
#pragma unroll
        for (int r = 0; r < 8; ++r)
            lrow[r] = lrow[r] * scl[r] + rowsum16(pvv[0][r] + pvv[1][r]);
#pragma unroll
        for (int t = 0; t < 4; ++t)
#pragma unroll
            for (int r = 0; r < 8; ++r) o[t][r] *= scl[r];

        __syncthreads();           // P stores visible before fragment reads
        // ---- P (16x32) @ V (32x64)   (4 WMMA)
        const __bf16* pr = &Pl[lr][0];
        bf16x16 pfrag = load_frag(pr + half * 16, pr + half * 16 + 8);
        const int jbase = jt * 16;
        int colb = jbase + half * 16; if (colb > NSEQ - 16) colb = NSEQ - 16;
#pragma unroll
        for (int nt = 0; nt < 4; ++nt) {
            const __bf16* vr = vb + (size_t)(nt * 16 + lr) * NSEQ + colb;
            o[nt] = wmma_bf16(pfrag, load_frag(vr, vr + 8), o[nt]);
        }
        __syncthreads();
    }

    // ---- epilogue: normalize and write attn output [B*N, 512] bf16
    const int b = bh >> 3, hh = bh & 7;
#pragma unroll
    for (int r = 0; r < 8; ++r) {
        const float inv = 1.f / lrow[r];
        const int ii = half * 8 + r;
        const size_t row = ((size_t)b * NSEQ + i0 + ii) * DMODEL + hh * DHEAD;
#pragma unroll
        for (int nt = 0; nt < 4; ++nt)
            ao[row + nt * 16 + lr] = f2bf(o[nt][r] * inv);
    }
}

// ---------------------------------------------------------------- out GEMM
// Same 32x32-per-wave register blocking; stores f32 directly to d_out.
__global__ __launch_bounds__(256) void out_gemm_kernel(
    const __bf16* __restrict__ ao, const __bf16* __restrict__ woT,
    float* __restrict__ out) {
    const int lane = threadIdx.x & 31, half = lane >> 4, lr = lane & 15;
    const int wave = threadIdx.x >> 5;
    const int mt = blockIdx.x;                 // 0..127
    const int nt = blockIdx.y * 8 + wave;      // 0..15
    const __bf16* ar0 = ao + (size_t)(mt * 32 + lr) * DMODEL;
    const __bf16* ar1 = ar0 + (size_t)16 * DMODEL;
    const __bf16* br0 = woT + (size_t)(nt * 32 + lr) * DMODEL;
    const __bf16* br1 = br0 + (size_t)16 * DMODEL;

    f32x8 c00 = zero8(), c01 = zero8(), c10 = zero8(), c11 = zero8();
#pragma unroll
    for (int ks = 0; ks < DMODEL / 32; ++ks) {
        const int dd = ks * 32;
        bf16x16 a0 = load_frag(ar0 + dd + half * 16, ar0 + dd + half * 16 + 8);
        bf16x16 a1 = load_frag(ar1 + dd + half * 16, ar1 + dd + half * 16 + 8);
        bf16x16 b0 = load_frag(br0 + dd + half * 16, br0 + dd + half * 16 + 8);
        bf16x16 b1 = load_frag(br1 + dd + half * 16, br1 + dd + half * 16 + 8);
        c00 = wmma_bf16(a0, b0, c00);
        c01 = wmma_bf16(a0, b1, c01);
        c10 = wmma_bf16(a1, b0, c10);
        c11 = wmma_bf16(a1, b1, c11);
    }
    f32x8 cs[2][2] = {{c00, c01}, {c10, c11}};
#pragma unroll
    for (int mi = 0; mi < 2; ++mi)
#pragma unroll
        for (int ni = 0; ni < 2; ++ni)
#pragma unroll
            for (int r = 0; r < 8; ++r)
                out[(size_t)(mt * 32 + mi * 16 + half * 8 + r) * DMODEL +
                    nt * 32 + ni * 16 + lr] = cs[mi][ni][r];
}

// ---------------------------------------------------------------- launcher
extern "C" void kernel_launch(void* const* d_in, const int* in_sizes, int n_in,
                              void* d_out, int out_size, void* d_ws, size_t ws_size,
                              hipStream_t stream) {
    (void)in_sizes; (void)n_in; (void)out_size; (void)ws_size;
    const float* x  = (const float*)d_in[0];
    const float* Wq = (const float*)d_in[1];
    const float* Wk = (const float*)d_in[2];
    const float* Wv = (const float*)d_in[3];
    const float* Wo = (const float*)d_in[4];
    const float* Er = (const float*)d_in[5];
    float* out = (float*)d_out;

    char* ws = (char*)d_ws;
    size_t off = 0;
    auto alloc = [&](size_t bytes) {
        void* p = ws + off;
        off += (bytes + 255) & ~(size_t)255;
        return p;
    };
    __bf16* xb  = (__bf16*)alloc((size_t)MTOK * DMODEL * 2);
    __bf16* wqT = (__bf16*)alloc((size_t)DMODEL * DMODEL * 2);
    __bf16* wkT = (__bf16*)alloc((size_t)DMODEL * DMODEL * 2);
    __bf16* wvT = (__bf16*)alloc((size_t)DMODEL * DMODEL * 2);
    __bf16* woT = (__bf16*)alloc((size_t)DMODEL * DMODEL * 2);
    __bf16* erb = (__bf16*)alloc((size_t)NSEQ * DHEAD * 2);
    __bf16* qg  = (__bf16*)alloc((size_t)MTOK * DMODEL * 2);
    __bf16* kg  = (__bf16*)alloc((size_t)MTOK * DMODEL * 2);
    __bf16* vtg = (__bf16*)alloc((size_t)MTOK * DMODEL * 2);
    __bf16* aog = (__bf16*)alloc((size_t)MTOK * DMODEL * 2);

    conv_kernel<<<2048, 256, 0, stream>>>(x, xb, MTOK * DMODEL);
    conv_kernel<<<512, 256, 0, stream>>>(Er, erb, NSEQ * DHEAD);
    convT_kernel<<<dim3(DMODEL / 16, DMODEL / 16, 4), 256, 0, stream>>>(
        Wq, Wk, Wv, Wo, wqT, wkT, wvT, woT);
    qkv_gemm_kernel<<<dim3(MTOK / 32, 2, 3), 256, 0, stream>>>(
        xb, wqT, wkT, wvT, qg, kg, vtg);
    attn_kernel<<<dim3(NSEQ / 16, BDIM * NHEADS), 32, 0, stream>>>(
        qg, kg, vtg, erb, aog);
    out_gemm_kernel<<<dim3(MTOK / 32, 2), 256, 0, stream>>>(aog, woT, out);
}
